// MSAGE_13013750907565
// MI455X (gfx1250) — matrix operand order repaired and checked by reading, not compiled, not run on hardware
//
#include <hip/hip_runtime.h>
#include <hip/hip_bf16.h>

#define NN   16384
#define INF_ 128
#define FH   512
#define RR   3
#define KK   16

typedef __attribute__((ext_vector_type(16))) __bf16 v16bf;
typedef __attribute__((ext_vector_type(8)))  __bf16 v8bf;
typedef __attribute__((ext_vector_type(8)))  float  v8f;

__device__ __forceinline__ v8f wmma_bf16(v16bf a, v16bf b, v8f c) {
    return __builtin_amdgcn_wmma_f32_16x16x32_bf16(false, a, false, b, (short)0, c,
                                                   false, false);
}
__device__ __forceinline__ int c_row(int lane, int v) { return v + ((lane >> 4) << 3); }
__device__ __forceinline__ int c_col(int lane) { return lane & 15; }

// A fragment (16x32 bf16, M x K) from an LDS tile laid out [16][128] row-major.
// ISA 7.12.2: lanes 0-15 -> M=lane, K = kbase + {0..7, 16..23};
//             lanes 16-31 -> M=lane-16, K = kbase + {8..15, 24..31}.
__device__ __forceinline__ v16bf load_A16(const __bf16* base, int lane, int kbase) {
    int m   = lane & 15;
    int off = (lane < 16) ? 0 : 8;
    union { v16bf v; v8bf h[2]; } u;
    u.h[0] = *(const v8bf*)(base + m * INF_ + kbase + off);
    u.h[1] = *(const v8bf*)(base + m * INF_ + kbase + 16 + off);
    return u.v;
}
// B fragment (32x16 bf16, K x N) from LDS weights stored [rows][128] row-major,
// row index = output column n (i.e. B[k][n] = W[n][k]).
// Lanes 0-15: N=lane, K = kbase + {0..7, 16..23}; lanes 16-31: N=lane-16, K = kbase + {8..15, 24..31}.
__device__ __forceinline__ v16bf load_B16(const __bf16* w, int ncol, int lane, int kbase) {
    int n   = ncol + (lane & 15);
    int off = (lane < 16) ? 0 : 8;
    union { v16bf v; v8bf h[2]; } u;
    u.h[0] = *(const v8bf*)(w + n * INF_ + kbase + off);
    u.h[1] = *(const v8bf*)(w + n * INF_ + kbase + 16 + off);
    return u.v;
}
__device__ __forceinline__ float sigmoidf_(float x) { return 1.f / (1.f + __expf(-x)); }

// ---------------------------------------------------------------- cvt: x -> bf16
__global__ void __launch_bounds__(256) cvt_bf16_kernel(const float* __restrict__ x,
                                                       __bf16* __restrict__ xb) {
    int i = blockIdx.x * 256 + threadIdx.x;
    xb[i] = (__bf16)x[i];
}

// ------------------------------------------- GX[n,:] = x[n,:] @ W_ih[r].T + b_lstm[r]
__global__ void __launch_bounds__(256) gx_kernel(const __bf16* __restrict__ xb,
                                                 const float* __restrict__ W_ih,
                                                 const float* __restrict__ b_lstm,
                                                 __bf16* __restrict__ GX, int r) {
    __shared__ __align__(16) __bf16 Wlds[FH * INF_];   // 128 KB
    __shared__ __align__(16) __bf16 xrow[16 * INF_];   // 4 KB
    __shared__ float blds[FH];
    int tid = threadIdx.x, lane = tid & 31, wave = tid >> 5;
    int n0 = blockIdx.x * 16;

    const float* Wg = W_ih + (size_t)r * FH * INF_;
    for (int i = tid * 4; i < FH * INF_; i += 1024) {
        float4 v = *(const float4*)(Wg + i);
        Wlds[i] = (__bf16)v.x; Wlds[i + 1] = (__bf16)v.y;
        Wlds[i + 2] = (__bf16)v.z; Wlds[i + 3] = (__bf16)v.w;
    }
    for (int i = tid; i < FH; i += 256) blds[i] = b_lstm[r * FH + i];
    for (int i = tid; i < 16 * INF_; i += 256) xrow[i] = xb[(size_t)n0 * INF_ + i];
    __syncthreads();

    v16bf af[4];
#pragma unroll
    for (int kc = 0; kc < 4; ++kc) af[kc] = load_A16(xrow, lane, kc * 32);
#pragma unroll
    for (int i = 0; i < 4; ++i) {
        int nt = wave * 4 + i;
        float bcol = blds[nt * 16 + c_col(lane)];
        v8f acc;
#pragma unroll
        for (int v = 0; v < 8; ++v) acc[v] = bcol;
#pragma unroll
        for (int kc = 0; kc < 4; ++kc)
            acc = wmma_bf16(af[kc], load_B16(Wlds, nt * 16, lane, kc * 32), acc);
#pragma unroll
        for (int v = 0; v < 8; ++v)
            GX[(size_t)(n0 + c_row(lane, v)) * FH + nt * 16 + c_col(lane)] = (__bf16)acc[v];
    }
}

// ------------------- per-relation LSTM recurrence + fused fc_self/fc_neigh epilogue
__global__ void __launch_bounds__(256) lstm_kernel(const __bf16* __restrict__ GX,
                                                   const __bf16* __restrict__ xb,
                                                   const int* __restrict__ nbr,
                                                   const float* __restrict__ W_hh,
                                                   const float* __restrict__ fc_self,
                                                   const float* __restrict__ fc_neigh,
                                                   const float* __restrict__ bias,
                                                   float* __restrict__ Xbuf, int r) {
    __shared__ __align__(16) __bf16 Wlds[FH * INF_];  // 128 KB: W_hh, later fc weights
    __shared__ __align__(16) __bf16 gxl[16 * FH];     // 16 KB: gathered gate rows / x rows
    __shared__ float gates[16 * FH];                  // 32 KB
    __shared__ __align__(16) __bf16 hl[16 * INF_];    // 4 KB
    int tid = threadIdx.x, lane = tid & 31, wave = tid >> 5;
    int n0 = blockIdx.x * 16;

    const float* Wg = W_hh + (size_t)r * FH * INF_;
    for (int i = tid * 4; i < FH * INF_; i += 1024) {
        float4 v = *(const float4*)(Wg + i);
        Wlds[i] = (__bf16)v.x; Wlds[i + 1] = (__bf16)v.y;
        Wlds[i + 2] = (__bf16)v.z; Wlds[i + 3] = (__bf16)v.w;
    }
    int em = tid >> 4;         // node 0..15 owned for elementwise phase
    int ej = (tid & 15) * 8;   // hidden-unit base
    float cst[8];
#pragma unroll
    for (int e = 0; e < 8; ++e) { cst[e] = 0.f; hl[em * INF_ + ej + e] = (__bf16)0.f; }
    __syncthreads();

    for (int t = 0; t < KK; ++t) {
        {   // gather precomputed gate rows GX[nbr_idx[r][n][t]] into LDS (1 KB/row)
            int m = tid >> 4, part = tid & 15;
            int src = nbr[((size_t)r * NN + (n0 + m)) * KK + t];
            const uint4* s4 = (const uint4*)(GX + (size_t)src * FH);
            uint4* d4 = (uint4*)(gxl + m * FH);
#pragma unroll
            for (int q = 0; q < 4; ++q) d4[part * 4 + q] = s4[part * 4 + q];
        }
        __syncthreads();

        v16bf af[4];
#pragma unroll
        for (int kc = 0; kc < 4; ++kc) af[kc] = load_A16(hl, lane, kc * 32);
#pragma unroll
        for (int i = 0; i < 4; ++i) {
            int nt = wave * 4 + i;
            v8f acc;
#pragma unroll
            for (int v = 0; v < 8; ++v)
                acc[v] = (float)gxl[c_row(lane, v) * FH + nt * 16 + c_col(lane)];
#pragma unroll
            for (int kc = 0; kc < 4; ++kc)
                acc = wmma_bf16(af[kc], load_B16(Wlds, nt * 16, lane, kc * 32), acc);
#pragma unroll
            for (int v = 0; v < 8; ++v)
                gates[c_row(lane, v) * FH + nt * 16 + c_col(lane)] = acc[v];
        }
        __syncthreads();

#pragma unroll
        for (int e = 0; e < 8; ++e) {   // i,f,g,o -> c,h   (PyTorch gate order)
            int j = ej + e;
            float ig = sigmoidf_(gates[em * FH + j]);
            float fg = sigmoidf_(gates[em * FH + 128 + j]);
            float gg = tanhf(gates[em * FH + 256 + j]);
            float og = sigmoidf_(gates[em * FH + 384 + j]);
            float c = fg * cst[e] + ig * gg;
            cst[e] = c;
            hl[em * INF_ + j] = (__bf16)(og * tanhf(c));
        }
        __syncthreads();
    }

    // ---- epilogue: X[:,r,:] = leaky_relu(x @ fc_self.T + h @ fc_neigh.T + bias)
    const float* fs = fc_self + (size_t)r * INF_ * INF_;
    const float* fn = fc_neigh + (size_t)r * INF_ * INF_;
    for (int i = tid * 4; i < INF_ * INF_; i += 1024) {
        float4 v = *(const float4*)(fs + i);
        Wlds[i] = (__bf16)v.x; Wlds[i + 1] = (__bf16)v.y;
        Wlds[i + 2] = (__bf16)v.z; Wlds[i + 3] = (__bf16)v.w;
        float4 u = *(const float4*)(fn + i);
        Wlds[16384 + i] = (__bf16)u.x; Wlds[16384 + i + 1] = (__bf16)u.y;
        Wlds[16384 + i + 2] = (__bf16)u.z; Wlds[16384 + i + 3] = (__bf16)u.w;
    }
    for (int i = tid; i < 16 * INF_; i += 256) gxl[i] = xb[(size_t)n0 * INF_ + i];
    __syncthreads();

    int nt = wave;   // 8 waves cover the 128 outputs
    float bcol = bias[r * INF_ + nt * 16 + c_col(lane)];
    v8f acc;
#pragma unroll
    for (int v = 0; v < 8; ++v) acc[v] = bcol;
#pragma unroll
    for (int kc = 0; kc < 4; ++kc) {
        acc = wmma_bf16(load_A16(gxl, lane, kc * 32), load_B16(Wlds, nt * 16, lane, kc * 32), acc);
        acc = wmma_bf16(load_A16(hl, lane, kc * 32), load_B16(Wlds + 16384, nt * 16, lane, kc * 32), acc);
    }
#pragma unroll
    for (int v = 0; v < 8; ++v) {
        float val = acc[v];
        val = val > 0.f ? val : 0.01f * val;
        Xbuf[((size_t)(n0 + c_row(lane, v)) * RR + r) * INF_ + nt * 16 + c_col(lane)] = val;
    }
}

// ---------------------------------------------------- tc = conn @ trans.T  [N,128]
__global__ void __launch_bounds__(256) tc_kernel(const float* __restrict__ conn,
                                                 const float* __restrict__ trans,
                                                 float* __restrict__ tc) {
    __shared__ __align__(16) __bf16 Tl[INF_ * INF_];  // 32 KB
    __shared__ __align__(16) __bf16 cl[16 * INF_];    // 4 KB
    int tid = threadIdx.x, lane = tid & 31, wave = tid >> 5;
    int n0 = blockIdx.x * 16;
    for (int i = tid * 4; i < INF_ * INF_; i += 1024) {
        float4 v = *(const float4*)(trans + i);
        Tl[i] = (__bf16)v.x; Tl[i + 1] = (__bf16)v.y;
        Tl[i + 2] = (__bf16)v.z; Tl[i + 3] = (__bf16)v.w;
    }
    for (int i = tid; i < 16 * INF_; i += 256) cl[i] = (__bf16)conn[(size_t)n0 * INF_ + i];
    __syncthreads();

    int nt = wave;
    v8f acc;
#pragma unroll
    for (int v = 0; v < 8; ++v) acc[v] = 0.f;
#pragma unroll
    for (int kc = 0; kc < 4; ++kc)
        acc = wmma_bf16(load_A16(cl, lane, kc * 32), load_B16(Tl, nt * 16, lane, kc * 32), acc);
#pragma unroll
    for (int v = 0; v < 8; ++v)
        tc[(size_t)(n0 + c_row(lane, v)) * INF_ + nt * 16 + c_col(lane)] = acc[v];
}

// --------------------------- per-node 3-way attention softmax + weighted combine
__global__ void __launch_bounds__(256) attn_kernel(const float* __restrict__ Xbuf,
                                                   const float* __restrict__ tc,
                                                   float* __restrict__ out) {
    int tid = threadIdx.x, lane = tid & 31, wave = tid >> 5;
    int n = blockIdx.x * 8 + wave;
    float tcv[4], xv[3][4], s[3];
#pragma unroll
    for (int q = 0; q < 4; ++q) tcv[q] = tc[(size_t)n * INF_ + lane + q * 32];
#pragma unroll
    for (int r = 0; r < 3; ++r) {
        float d = 0.f;
#pragma unroll
        for (int q = 0; q < 4; ++q) {
            xv[r][q] = Xbuf[((size_t)n * RR + r) * INF_ + lane + q * 32];
            d += xv[r][q] * tcv[q];
        }
#pragma unroll
        for (int off = 16; off > 0; off >>= 1) d += __shfl_xor(d, off, 32);
        s[r] = d;
    }
    float mx = fmaxf(s[0], fmaxf(s[1], s[2]));
    float e0 = __expf(s[0] - mx), e1 = __expf(s[1] - mx), e2 = __expf(s[2] - mx);
    float inv = 1.f / (e0 + e1 + e2);
    e0 *= inv; e1 *= inv; e2 *= inv;
#pragma unroll
    for (int q = 0; q < 4; ++q)
        out[(size_t)n * INF_ + lane + q * 32] =
            e0 * xv[0][q] + e1 * xv[1][q] + e2 * xv[2][q];
}

extern "C" void kernel_launch(void* const* d_in, const int* in_sizes, int n_in,
                              void* d_out, int out_size, void* d_ws, size_t ws_size,
                              hipStream_t stream) {
    const float* x        = (const float*)d_in[0];
    const float* conn     = (const float*)d_in[1];
    const int*   nbr      = (const int*)d_in[2];
    const float* trans    = (const float*)d_in[3];
    const float* W_ih     = (const float*)d_in[4];
    const float* W_hh     = (const float*)d_in[5];
    const float* b_lstm   = (const float*)d_in[6];
    const float* fc_self  = (const float*)d_in[7];
    const float* fc_neigh = (const float*)d_in[8];
    const float* bias     = (const float*)d_in[9];
    float* out = (float*)d_out;

    // workspace layout: xb 4MB | GX 16MB (reused per relation) | Xbuf 24MB | tc 8MB
    char* ws = (char*)d_ws;
    __bf16* xb   = (__bf16*)ws;
    __bf16* GX   = (__bf16*)(ws + (4 << 20));
    float*  Xbuf = (float*)(ws + (20 << 20));
    float*  tcb  = (float*)(ws + (44 << 20));

    cvt_bf16_kernel<<<NN * INF_ / 256, 256, 0, stream>>>(x, xb);
    for (int r = 0; r < RR; ++r) {
        gx_kernel<<<NN / 16, 256, 0, stream>>>(xb, W_ih, b_lstm, GX, r);
        lstm_kernel<<<NN / 16, 256, 0, stream>>>(GX, xb, nbr, W_hh, fc_self, fc_neigh,
                                                 bias, Xbuf, r);
    }
    tc_kernel<<<NN / 16, 256, 0, stream>>>(conn, trans, tcb);
    attn_kernel<<<NN / 8, 256, 0, stream>>>(Xbuf, tcb, out);
    (void)in_sizes; (void)n_in; (void)out_size; (void)ws_size;
}
